// Mamba4KT_47785806135740
// MI455X (gfx1250) — matrix operand order repaired and checked by reading, not compiled, not run on hardware
//
#include <hip/hip_runtime.h>
#include <hip/hip_bf16.h>
#include <math.h>

typedef __attribute__((ext_vector_type(16))) _Float16 v16h;
typedef __attribute__((ext_vector_type(8)))  _Float16 v8h;
typedef __attribute__((ext_vector_type(8)))  float    v8f;

#define NTOK   32768   // 64 * 512
#define SEQLEN 513
#define DM     256
#define DI     512
#define DSTATE 16
#define DTRANK 16

__device__ __forceinline__ float sigmoidf_(float x){ return 1.f/(1.f+expf(-x)); }
__device__ __forceinline__ float siluf_(float x){ return x*sigmoidf_(x); }

// ---------------- staging / conversion ----------------
__global__ void k_f32_to_f16(const float* __restrict__ s, _Float16* __restrict__ d, int n){
  int i = blockIdx.x*256 + threadIdx.x;
  if (i < n) d[i] = (_Float16)s[i];
}

// dt_w (512x16) -> f16 (512x32), cols 16..31 zero (K padded for WMMA)
__global__ void k_pad_dtw(const float* __restrict__ s, _Float16* __restrict__ d){
  int i = blockIdx.x*256 + threadIdx.x;
  if (i < DI*32){ int r = i>>5, c = i&31;
    d[i] = (c<DTRANK) ? (_Float16)s[r*DTRANK+c] : (_Float16)0.f; }
}

// ---------------- embedding + reg_loss partials ----------------
__global__ void k_embed(const int* __restrict__ q, const int* __restrict__ c, const int* __restrict__ r,
                        const float* __restrict__ concept_enc, const float* __restrict__ answer_enc,
                        const float* __restrict__ q_diff, const float* __restrict__ a_diff,
                        float* __restrict__ x32, _Float16* __restrict__ x16,
                        float* __restrict__ reg_partial){
  __shared__ float red[256];
  int tok = blockIdx.x; int b = tok>>9, t = tok&511;
  int ii = b*SEQLEN + t;
  int qi = q[ii], ci = c[ii], ri = r[ii];
  int rm = (ri > -1) ? ri : 0;           // responses * (responses > -1)
  int d = threadIdx.x;
  float qd = q_diff[(size_t)qi*DM + d];
  float v = answer_enc[(size_t)rm*DM + d] + concept_enc[(size_t)ci*DM + d]
          + qd * a_diff[(size_t)rm*DM + d];
  size_t o = (size_t)tok*DM + d;
  x32[o] = v; x16[o] = (_Float16)v;
  red[d] = qd*qd; __syncthreads();
  for (int s=128; s>0; s>>=1){ if (d<s) red[d]+=red[d+s]; __syncthreads(); }
  if (d==0) reg_partial[tok] = red[0];
}

__global__ void k_reduce_reg(const float* __restrict__ partial, float* __restrict__ out){
  __shared__ float red[256];
  int d = threadIdx.x;
  float s = 0.f;
  for (int i=d; i<NTOK; i+=256) s += partial[i];   // fixed order -> deterministic
  red[d]=s; __syncthreads();
  for (int t=128; t>0; t>>=1){ if (d<t) red[d]+=red[d+t]; __syncthreads(); }
  if (d==0) *out = red[0];
}

__global__ void k_rshft(const int* __restrict__ r, float* __restrict__ out){
  int i = blockIdx.x*256 + threadIdx.x;
  if (i < NTOK){ int b=i>>9, t=i&511; out[i] = (float)r[b*SEQLEN + t + 1]; }
}

// ---------------- WMMA GEMM:  C[MxN] f32 = A[MxK] f16 @ W[NxK]^T f16 ----------------
// One wave computes a 16 x (16*JT) output strip; JT is compile-time so the
// inner loop is branch-free (EXEC stays all-ones around every WMMA) and the
// grid is launched exactly (no early-exit, no saveexec, no spills).
template<int JT>
__global__ __launch_bounds__(128) void k_wmma_gemm_nt(const _Float16* __restrict__ A,
                               const _Float16* __restrict__ W,
                               float* __restrict__ C, int N, int K){
  const int lane = threadIdx.x & 31;
  const int wave = blockIdx.x*4 + (threadIdx.x>>5);
  const int ngrp = N/(16*JT);
  const int mg = wave / ngrp;
  const int ng = wave % ngrp;
  const int lm   = lane & 15;
  const int khi  = (lane & 16) ? 16 : 0;       // A K offset     (contiguous k = 16h + s)
  const int koff = (lane & 16) ? 16 : 0;       // B K offset     (contiguous k = 16h + s)
  const _Float16* ap = A + ((size_t)(mg*16 + lm))*(size_t)K + khi;
  const _Float16* wb = W + ((size_t)(ng*16*JT + lm))*(size_t)K + koff;
  v8f acc[JT] = {};
  for (int k0=0; k0<K; k0+=32){
    v8h lo = *(const v8h*)(ap + k0);
    v8h hi = *(const v8h*)(ap + k0 + 8);
    v16h a = __builtin_shufflevector(lo,hi,0,1,2,3,4,5,6,7,8,9,10,11,12,13,14,15);
    #pragma unroll
    for (int j=0;j<JT;j++){
      v16h bf = *(const v16h*)(wb + (size_t)(j*16)*(size_t)K + k0);
      acc[j] = __builtin_amdgcn_wmma_f32_16x16x32_f16(
                 false, a, false, bf, (short)0, acc[j], false, false);
    }
  }
  const int rb = mg*16 + ((lane&16)?8:0);
  float* cb = C + (size_t)rb*(size_t)N + ng*16*JT + lm;
  #pragma unroll
  for (int j=0;j<JT;j++){
    #pragma unroll
    for (int r2=0;r2<8;r2++)
      cb[(size_t)r2*(size_t)N + j*16] = acc[j][r2];
  }
}

// ---------------- causal depthwise conv(4) + SiLU ----------------
__global__ void k_conv_silu(const float* __restrict__ xz, const float* __restrict__ cw,
                            const float* __restrict__ cb, float* __restrict__ xb32,
                            _Float16* __restrict__ xb16){
  int tok = blockIdx.x; int b = tok>>9, t = tok&511;
  int d = threadIdx.x;                          // blockDim = 512
  float s = cb[d];
  #pragma unroll
  for (int k=0;k<4;k++){
    int tt = t-3+k;
    if (tt>=0) s += cw[d*4+k] * xz[((size_t)(b*512+tt))*1024 + d];
  }
  s = siluf_(s);
  size_t o = (size_t)tok*DI + d;
  xb32[o]=s; xb16[o]=(_Float16)s;
}

// xdbl[:, :16] -> f16 Nx32 (zero-padded K for WMMA)
__global__ void k_stage_dtin(const float* __restrict__ xdbl, _Float16* __restrict__ dtin){
  int i = blockIdx.x*256 + threadIdx.x;
  if (i < NTOK*32){ int tok=i>>5, c=i&31;
    dtin[i] = (c<DTRANK) ? (_Float16)xdbl[(size_t)tok*48 + c] : (_Float16)0.f; }
}

__global__ void k_softplus(float* __restrict__ dt, const float* __restrict__ dtb){
  int i = blockIdx.x*256 + threadIdx.x;
  if (i < NTOK*DI){ int d = i & (DI-1);
    float x = dt[i] + dtb[d];
    dt[i] = (x > 20.f) ? x : log1pf(expf(x)); }
}

// ---------------- selective scan (fused with +xb*D and *silu(z), writes f16 for next GEMM) ----------------
__global__ void k_scan(const float* __restrict__ dt, const float* __restrict__ xb,
                       const float* __restrict__ xdbl, const float* __restrict__ xz,
                       const float* __restrict__ A_log, const float* __restrict__ Dp,
                       _Float16* __restrict__ y16){
  __shared__ float sBC[32];                     // B[16] then C[16]
  int b = blockIdx.x;                           // 64 blocks
  int d = threadIdx.x;                          // 512 channels
  float A[DSTATE];
  #pragma unroll
  for (int s=0;s<DSTATE;s++) A[s] = -expf(A_log[d*DSTATE+s]);
  float Dv = Dp[d];
  float h[DSTATE];
  #pragma unroll
  for (int s=0;s<DSTATE;s++) h[s] = 0.f;
  for (int t=0;t<512;t++){
    int tok = b*512 + t;
    if (threadIdx.x < 32) sBC[threadIdx.x] = xdbl[(size_t)tok*48 + 16 + threadIdx.x];
    __syncthreads();
    float dtv = dt[(size_t)tok*DI + d];
    float xv  = xb[(size_t)tok*DI + d];
    float dx  = dtv*xv;
    float y = 0.f;
    #pragma unroll
    for (int s=0;s<DSTATE;s++){
      float dA = expf(dtv*A[s]);
      h[s] = dA*h[s] + dx*sBC[s];
      y += h[s]*sBC[16+s];
    }
    float z = xz[(size_t)tok*1024 + DI + d];
    float out = (y + xv*Dv) * siluf_(z);
    y16[(size_t)tok*DI + d] = (_Float16)out;
    __syncthreads();
  }
}

// ---------------- residual(+bias) + RMSNorm, writes f32 + f16 ----------------
__global__ void k_add_rmsnorm(const float* __restrict__ a, const float* __restrict__ b,
                              const float* __restrict__ bias, const float* __restrict__ w,
                              float* __restrict__ o32, _Float16* __restrict__ o16){
  __shared__ float red[256];
  int tok = blockIdx.x, d = threadIdx.x;
  size_t i = (size_t)tok*DM + d;
  float v = a[i] + b[i];
  if (bias) v += bias[d];
  red[d] = v*v; __syncthreads();
  for (int s=128; s>0; s>>=1){ if (d<s) red[d]+=red[d+s]; __syncthreads(); }
  float r = rsqrtf(red[0]*(1.f/DM) + 1e-12f);
  float o = v*w[d]*r;
  o32[i] = o; o16[i] = (_Float16)o;
}

__global__ void k_bias_gelu(const float* __restrict__ s, const float* __restrict__ b1,
                            _Float16* __restrict__ hid16){
  int i = blockIdx.x*256 + threadIdx.x;
  if (i < NTOK*1024){
    float v = s[i] + b1[i & 1023];
    float g = 0.5f*v*(1.f + erff(v*0.70710678118654752f));  // exact GeLU
    hid16[i] = (_Float16)g;
  }
}

// ---------------- gathered logit: pred[tok] = sigmoid(h . out_w[cshft] + out_b[cshft]) ----------------
__global__ void k_pred(const float* __restrict__ h, const float* __restrict__ ow,
                       const float* __restrict__ ob, const int* __restrict__ skills,
                       float* __restrict__ pred){
  int wid  = (blockIdx.x*blockDim.x + threadIdx.x) >> 5;  // one wave per token
  int lane = threadIdx.x & 31;
  int b = wid>>9, t = wid&511;
  int c = skills[b*SEQLEN + t + 1];
  const float* hr = h  + (size_t)wid*DM;
  const float* wr = ow + (size_t)c*DM;
  float s = 0.f;
  for (int i=lane; i<DM; i+=32) s += hr[i]*wr[i];
  #pragma unroll
  for (int off=16; off>0; off>>=1) s += __shfl_down(s, off, 32);
  if (lane==0) pred[wid] = sigmoidf_(s + ob[c]);
}

extern "C" void kernel_launch(void* const* d_in, const int* in_sizes, int n_in,
                              void* d_out, int out_size, void* d_ws, size_t ws_size,
                              hipStream_t stream) {
  (void)in_sizes; (void)n_in; (void)out_size; (void)ws_size;
  const int*   questions   = (const int*)  d_in[0];
  const int*   skills      = (const int*)  d_in[1];
  const int*   responses   = (const int*)  d_in[2];
  const float* concept_enc = (const float*)d_in[3];
  const float* answer_enc  = (const float*)d_in[4];
  const float* q_diff      = (const float*)d_in[5];
  /* d_in[6] = c_diff, unused by forward */
  const float* a_diff      = (const float*)d_in[7];
  const float* out_w       = (const float*)d_in[8];
  const float* out_b       = (const float*)d_in[9];

  struct Blk { const float *in_proj,*conv_w,*conv_b,*x_proj,*dt_w,*dt_b,*A_log,*D,
                           *out_proj,*ln_w,*w1,*b1,*w2,*b2,*ffn_ln; } blk[2];
  for (int i=0;i<2;i++){
    void* const* p = d_in + 10 + i*15;
    blk[i].in_proj=(const float*)p[0];  blk[i].conv_w=(const float*)p[1];
    blk[i].conv_b =(const float*)p[2];  blk[i].x_proj=(const float*)p[3];
    blk[i].dt_w   =(const float*)p[4];  blk[i].dt_b  =(const float*)p[5];
    blk[i].A_log  =(const float*)p[6];  blk[i].D     =(const float*)p[7];
    blk[i].out_proj=(const float*)p[8]; blk[i].ln_w  =(const float*)p[9];
    blk[i].w1=(const float*)p[10]; blk[i].b1=(const float*)p[11];
    blk[i].w2=(const float*)p[12]; blk[i].b2=(const float*)p[13];
    blk[i].ffn_ln=(const float*)p[14];
  }

  float* pred_out  = (float*)d_out;
  float* rshft_out = pred_out + NTOK;
  float* reg_out   = pred_out + 2*NTOK;

  size_t off = 0;
  auto alloc = [&](size_t bytes)->void*{
    void* p = (char*)d_ws + off; off += (bytes + 255) & ~(size_t)255; return p; };
  float*    x32   = (float*)   alloc((size_t)NTOK*DM*4);
  _Float16* x16   = (_Float16*)alloc((size_t)NTOK*DM*2);
  float*    big32 = (float*)   alloc((size_t)NTOK*1024*4);  // xz, later FFN-hidden f32
  float*    xb32  = (float*)   alloc((size_t)NTOK*DI*4);
  _Float16* xb16  = (_Float16*)alloc((size_t)NTOK*DI*2);
  float*    xdbl  = (float*)   alloc((size_t)NTOK*48*4);
  _Float16* dtin  = (_Float16*)alloc((size_t)NTOK*32*2);
  float*    dt32  = (float*)   alloc((size_t)NTOK*DI*4);
  _Float16* y16   = (_Float16*)alloc((size_t)NTOK*DI*2);
  float*    o32   = (float*)   alloc((size_t)NTOK*DM*4);
  float*    h32   = (float*)   alloc((size_t)NTOK*DM*4);
  _Float16* h16   = (_Float16*)alloc((size_t)NTOK*DM*2);
  _Float16* hid16 = (_Float16*)alloc((size_t)NTOK*1024*2);
  float*    regp  = (float*)   alloc((size_t)NTOK*4);
  _Float16* wf16[2][6];
  const int wsz[6] = {1024*DM, 48*DI, DI*32, DM*DI, 1024*DM, DM*1024};
  for (int i=0;i<2;i++) for (int j=0;j<6;j++)
    wf16[i][j] = (_Float16*)alloc((size_t)wsz[j]*2);

  auto cvt = [&](const float* s, _Float16* d, int n){
    k_f32_to_f16<<<(n+255)/256, 256, 0, stream>>>(s, d, n); };
  auto gemm = [&](const _Float16* A, const _Float16* W, float* C, int M, int N, int K){
    if (N % 64 == 0){
      int waves = (M/16)*(N/64);                 // JT=4 strip = 16x64
      k_wmma_gemm_nt<4><<<waves/4, 128, 0, stream>>>(A, W, C, N, K);
    } else {                                     // N = 48 (x_proj)
      int waves = (M/16)*(N/48);                 // JT=3 strip = 16x48
      k_wmma_gemm_nt<3><<<waves/4, 128, 0, stream>>>(A, W, C, N, K);
    }
  };

  // weight staging (idempotent, deterministic)
  for (int i=0;i<2;i++){
    cvt(blk[i].in_proj,  wf16[i][0], 1024*DM);
    cvt(blk[i].x_proj,   wf16[i][1], 48*DI);
    k_pad_dtw<<<(DI*32+255)/256, 256, 0, stream>>>(blk[i].dt_w, wf16[i][2]);
    cvt(blk[i].out_proj, wf16[i][3], DM*DI);
    cvt(blk[i].w1,       wf16[i][4], 1024*DM);
    cvt(blk[i].w2,       wf16[i][5], DM*1024);
  }

  k_embed<<<NTOK, 256, 0, stream>>>(questions, skills, responses,
      concept_enc, answer_enc, q_diff, a_diff, x32, x16, regp);
  k_reduce_reg<<<1, 256, 0, stream>>>(regp, reg_out);
  k_rshft<<<(NTOK+255)/256, 256, 0, stream>>>(responses, rshft_out);

  for (int i=0;i<2;i++){
    gemm(x16,  wf16[i][0], big32, NTOK, 1024, DM);             // xz = x @ in_proj^T
    k_conv_silu<<<NTOK, DI, 0, stream>>>(big32, blk[i].conv_w, blk[i].conv_b, xb32, xb16);
    gemm(xb16, wf16[i][1], xdbl,  NTOK, 48,   DI);             // xdbl = xb @ x_proj^T
    k_stage_dtin<<<(NTOK*32+255)/256, 256, 0, stream>>>(xdbl, dtin);
    gemm(dtin, wf16[i][2], dt32,  NTOK, DI,   32);             // dt raw (K padded 16->32)
    k_softplus<<<(NTOK*DI+255)/256, 256, 0, stream>>>(dt32, blk[i].dt_b);
    k_scan<<<64, DI, 0, stream>>>(dt32, xb32, xdbl, big32, blk[i].A_log, blk[i].D, y16);
    gemm(y16,  wf16[i][3], o32,   NTOK, DM,   DI);             // mamba_out
    k_add_rmsnorm<<<NTOK, DM, 0, stream>>>(o32, x32, nullptr, blk[i].ln_w, h32, h16);
    gemm(h16,  wf16[i][4], big32, NTOK, 1024, DM);             // ffn hidden raw
    k_bias_gelu<<<(NTOK*1024+255)/256, 256, 0, stream>>>(big32, blk[i].b1, hid16);
    gemm(hid16,wf16[i][5], o32,   NTOK, DM,   1024);           // ffn out raw
    k_add_rmsnorm<<<NTOK, DM, 0, stream>>>(o32, h32, blk[i].b2, blk[i].ffn_ln, x32, x16);
  }

  k_pred<<<NTOK/8, 256, 0, stream>>>(x32, out_w, out_b, skills, pred_out);
}
